// Equivariant_SecondOrder_Decoder_19069654794760
// MI455X (gfx1250) — matrix operand order repaired and checked by reading, not compiled, not run on hardware
//
#include <hip/hip_runtime.h>
#include <hip/hip_bf16.h>

// Problem constants (B, D_LAT==H, M, S)
constexpr int B_ = 8;
constexpr int D_ = 64;
constexpr int M_ = 384;
constexpr int S_ = 8;
constexpr int MT_ = M_ / 16;  // 24 tiles per side

typedef __bf16 bf16_t;
typedef __attribute__((ext_vector_type(16))) __bf16 v16bf;
typedef __attribute__((ext_vector_type(8)))  __bf16 v8bf;
typedef __attribute__((ext_vector_type(8)))  float  v8f;

// ---------------------------------------------------------------------------
// Small f32 side kernels: pooled statistics and basis projections.
// Activation tensors are bf16, layout [b][i][j][h] (h contiguous -> WMMA-friendly)
// ---------------------------------------------------------------------------

__global__ void k_meanP(const float* __restrict__ P, float* __restrict__ smean) {
  int b = blockIdx.x, d = threadIdx.x;
  const float* p = P + ((size_t)b * D_ + d) * M_;
  float acc = 0.f;
  for (int i = 0; i < M_; ++i) acc += p[i];
  smean[b * D_ + d] = acc / M_;
}

// layer_1_to_2 basis projections: rowv/colv/dvecv [b][h][i], constv [b][h]
__global__ void k_proj1(const float* __restrict__ P, const float* __restrict__ smean,
                        const float* __restrict__ c0, const float* __restrict__ bias0,
                        float* __restrict__ rowv, float* __restrict__ colv,
                        float* __restrict__ dvecv, float* __restrict__ constv) {
  int b = blockIdx.x / M_, i = blockIdx.x % M_, h = threadIdx.x;
  __shared__ float pl[D_], sl[D_];
  pl[h] = P[((size_t)b * D_ + h) * M_ + i];
  sl[h] = smean[b * D_ + h];
  __syncthreads();
  float rA = 0.f, cA = 0.f, dA = 0.f, kA = 0.f;
  for (int d = 0; d < D_; ++d) {
    const float* wp = c0 + ((size_t)d * D_ + h) * 5;
    float x = pl[d], s = sl[d];
    rA += wp[2] * x;             // op3 row
    cA += wp[3] * x;             // op4 col
    dA += wp[0] * x + wp[1] * s; // op1+op2 diag
    kA += wp[4] * s;             // op5 const
  }
  size_t o = ((size_t)b * D_ + h) * M_ + i;
  rowv[o] = rA; colv[o] = cA; dvecv[o] = dA;
  if (i == 0) constv[b * D_ + h] = kA + bias0[h];
}

// Materialize X1 = relu(layer_1_to_2(P)) as bf16 [b][i][j][h]
__global__ void k_mat1(const float* __restrict__ rowv, const float* __restrict__ colv,
                       const float* __restrict__ dvecv, const float* __restrict__ constv,
                       bf16_t* __restrict__ out) {
  int b = blockIdx.x / M_, i = blockIdx.x % M_;
  __shared__ float rv[D_], dv[D_], cb[D_];
  if (threadIdx.x < D_) {
    int h = threadIdx.x;
    rv[h] = rowv[((size_t)b * D_ + h) * M_ + i] + constv[b * D_ + h];
    dv[h] = dvecv[((size_t)b * D_ + h) * M_ + i];
  }
  __syncthreads();
  for (int j = threadIdx.x; j < M_; j += blockDim.x) {
    bf16_t* po = out + (((size_t)b * M_ + i) * M_ + j) * D_;
    const float dmask = (i == j) ? 1.f : 0.f;  // branchless diagonal select
    for (int hb = 0; hb < D_ / 8; ++hb) {
      v8bf o;
#pragma unroll
      for (int e = 0; e < 8; ++e) {
        int h = hb * 8 + e;
        float v = rv[h] + colv[((size_t)b * D_ + h) * M_ + j];
        v = fmaf(dv[h], dmask, v);
        o[e] = (bf16_t)fmaxf(v, 0.f);
      }
      *reinterpret_cast<v8bf*>(po + hb * 8) = o;
    }
  }
}

// Row sums (/M) + diagonal extraction: rows/diagv [b][i][d]
__global__ void k_rowStats(const bf16_t* __restrict__ X, float* __restrict__ rows,
                           float* __restrict__ diagv) {
  int b = blockIdx.x / M_, i = blockIdx.x % M_, d = threadIdx.x;
  const bf16_t* p = X + ((size_t)b * M_ + i) * M_ * D_ + d;
  float acc = 0.f, dg = 0.f;
  for (int j = 0; j < M_; ++j) {
    if (j + 64 < M_) __builtin_prefetch(&p[(size_t)(j + 64) * D_], 0, 0);
    float v = (float)p[(size_t)j * D_];
    acc += v;
    if (j == i) dg = v;
  }
  rows[((size_t)b * M_ + i) * D_ + d] = acc / M_;
  diagv[((size_t)b * M_ + i) * D_ + d] = dg;
}

// Column sums (/M): cols [b][j][d]
__global__ void k_colStats(const bf16_t* __restrict__ X, float* __restrict__ cols) {
  int b = blockIdx.x / M_, j = blockIdx.x % M_, d = threadIdx.x;
  const bf16_t* p = X + ((size_t)b * M_ * M_ + j) * D_ + d;
  float acc = 0.f;
  for (int i = 0; i < M_; ++i) {
    if (i + 8 < M_) __builtin_prefetch(&p[(size_t)(i + 8) * M_ * D_], 0, 0);
    acc += (float)p[(size_t)i * M_ * D_];
  }
  cols[((size_t)b * M_ + j) * D_ + d] = acc / M_;
}

// sum_all = sum_i rows[i]/M, sum_diag = sum_i diag[i]/M  -> [b][d]
__global__ void k_scalarStats(const float* __restrict__ rows, const float* __restrict__ diagv,
                              float* __restrict__ sall, float* __restrict__ sdiag) {
  int b = blockIdx.x, d = threadIdx.x;
  float sa = 0.f, sd = 0.f;
  for (int i = 0; i < M_; ++i) {
    size_t o = ((size_t)b * M_ + i) * D_ + d;
    sa += rows[o];
    sd += diagv[o];
  }
  sall[b * D_ + d] = sa / M_;
  sdiag[b * D_ + d] = sd / M_;
}

// layer_2_to_2 basis projections for ops 1-9,12-15 (everything but the full GEMMs)
__global__ void k_proj2(const float* __restrict__ rows, const float* __restrict__ cols,
                        const float* __restrict__ diagv, const float* __restrict__ sdiag,
                        const float* __restrict__ sall, const float* __restrict__ coeffs,
                        const float* __restrict__ diag_bias, const float* __restrict__ all_bias,
                        float* __restrict__ rowv, float* __restrict__ colv,
                        float* __restrict__ dvecv, float* __restrict__ constv) {
  int b = blockIdx.x / M_, i = blockIdx.x % M_, h = threadIdx.x;
  __shared__ float rl[D_], cl[D_], dl[D_], sdl[D_], sal[D_];
  size_t base = ((size_t)b * M_ + i) * D_;
  rl[h] = rows[base + h];
  cl[h] = cols[base + h];
  dl[h] = diagv[base + h];
  sdl[h] = sdiag[b * D_ + h];
  sal[h] = sall[b * D_ + h];
  __syncthreads();
  float rA = 0.f, cA = 0.f, dA = 0.f, kA = 0.f;
  for (int d = 0; d < D_; ++d) {
    const float* wp = coeffs + ((size_t)d * D_ + h) * 15;
    float r = rl[d], c = cl[d], dg = dl[d], sd = sdl[d], sa = sal[d];
    dA += wp[0] * dg + wp[1] * sd + wp[2] * r + wp[3] * c + wp[4] * sa;  // ops 1-5 (diag)
    rA += wp[5] * c + wp[6] * r + wp[11] * dg;                           // ops 6,7,12 (row)
    cA += wp[7] * c + wp[8] * r + wp[12] * dg;                           // ops 8,9,13 (col)
    kA += wp[13] * sd + wp[14] * sa;                                     // ops 14,15 (const)
  }
  size_t o = ((size_t)b * D_ + h) * M_ + i;
  rowv[o] = rA;
  colv[o] = cA;
  dvecv[o] = dA + diag_bias[h];
  if (i == 0) constv[b * D_ + h] = kA + all_bias[h];
}

// ---------------------------------------------------------------------------
// Main WMMA kernel: out[b,s,i,j] = relu( W9^T x(:,i,j) + W10^T x(:,j,i)
//                                        + row[s,i] + col[s,j] + const[s]
//                                        + (i==j)*dvec[s,i] )
// One block per (b, 16x16 tile). 8 wave32, 4 v_wmma_f32_16x16x32_bf16 per
// 16(s)x16(pixel) output tile. Epilogue terms staged [q][s] so each lane's
// 8 accumulator rows read as two contiguous ds_load_b128.
// ---------------------------------------------------------------------------
__global__ void __launch_bounds__(256, 1) k_wmma2to2(
    const bf16_t* __restrict__ Xin, const float* __restrict__ coeffs,
    const float* __restrict__ rowv, const float* __restrict__ colv,
    const float* __restrict__ dvecv, const float* __restrict__ constv,
    bf16_t* __restrict__ Xout) {
  const int j0 = blockIdx.x * 16, i0 = blockIdx.y * 16, b = blockIdx.z;
  __shared__ bf16_t WA[2][64][64];      // WA[t][s][d] = coeffs[d][s][9+t] (identity/transpose)
  __shared__ bf16_t XA[16][16][64];     // XA[ii][jj][d] = x[b, i0+ii, j0+jj, d]
  __shared__ bf16_t XT[16][16][64];     // XT[r][c][d]   = x[b, j0+r,  i0+c,  d]
  __shared__ __align__(32) float RS2[16][64];  // RS2[ii][s] = row[s,i0+ii] + const[s]
  __shared__ __align__(32) float CS2[16][64];  // CS2[jj][s] = col[s,j0+jj]
  __shared__ __align__(32) float DV2[16][64];  // DV2[ii][s] = dvec[s,i0+ii] (incl diag_bias)
  const int t = threadIdx.x;

  // stage weights as bf16
  for (int idx = t; idx < 2 * 64 * 64; idx += 256) {
    int tt = idx >> 12, rem = idx & 4095, s = rem >> 6, d = rem & 63;
    WA[tt][s][d] = (bf16_t)coeffs[((size_t)d * 64 + s) * 15 + 9 + tt];
  }
  // stage pooled terms for the epilogue (coalesced reads along q)
  for (int idx = t; idx < 64 * 16; idx += 256) {
    int s = idx >> 4, q = idx & 15;
    RS2[q][s] = rowv[((size_t)b * 64 + s) * M_ + i0 + q] + constv[b * 64 + s];
    CS2[q][s] = colv[((size_t)b * 64 + s) * M_ + j0 + q];
    DV2[q][s] = dvecv[((size_t)b * 64 + s) * M_ + i0 + q];
  }
  // stage x tiles (one 128B channel vector per thread per tile)
  {
    int ii = t >> 4, jj = t & 15;
    const uint4* s0 = (const uint4*)(Xin + (((size_t)b * M_ + i0 + ii) * M_ + (j0 + jj)) * 64);
    const uint4* s1 = (const uint4*)(Xin + (((size_t)b * M_ + j0 + ii) * M_ + (i0 + jj)) * 64);
    uint4* d0 = (uint4*)&XA[ii][jj][0];
    uint4* d1 = (uint4*)&XT[ii][jj][0];
#pragma unroll
    for (int k = 0; k < 8; ++k) { d0[k] = s0[k]; d1[k] = s1[k]; }
  }
  __syncthreads();

  const int wave = t >> 5, lane = t & 31, hi = lane >> 4, ln = lane & 15;
  const int sm = wave & 3;                 // s-block (4 blocks of 16 channels)
  const int ntBase = (wave >> 2) * 8;      // 8 i-rows per wave
  const int srow = sm * 16 + ln;
  const int sbase = sm * 16 + hi * 8;      // first accumulator channel for this lane

  // A fragments (16x32 bf16): lane<16 holds K {0..15}, lane>=16 holds K {16..31}
  v16bf A9a, A9b, A10a, A10b;
  {
    const bf16_t* p = &WA[0][srow][hi * 16];
#pragma unroll
    for (int e = 0; e < 8; ++e) { A9a[e] = p[e]; A9a[8 + e] = p[8 + e]; }
    p = &WA[0][srow][32 + hi * 16];
#pragma unroll
    for (int e = 0; e < 8; ++e) { A9b[e] = p[e]; A9b[8 + e] = p[8 + e]; }
    p = &WA[1][srow][hi * 16];
#pragma unroll
    for (int e = 0; e < 8; ++e) { A10a[e] = p[e]; A10a[8 + e] = p[8 + e]; }
    p = &WA[1][srow][32 + hi * 16];
#pragma unroll
    for (int e = 0; e < 8; ++e) { A10b[e] = p[e]; A10b[8 + e] = p[8 + e]; }
  }

  // column (j) epilogue term depends only on the lane -> hoist out of row loop
  const v8f c8 = *reinterpret_cast<const v8f*>(&CS2[ln][sbase]);

  for (int q = 0; q < 8; ++q) {
    const int nt = ntBase + q;  // output row i = i0 + nt; column j = j0 + ln
    // B fragments (32x16 bf16): lane = column, K split into low/high 16 (contiguous d)
    v16bf Bm0, Bm1, Tm0, Tm1;
    const bf16_t* pb = &XA[nt][ln][hi * 16];
#pragma unroll
    for (int e = 0; e < 16; ++e) Bm0[e] = pb[e];
    pb = &XA[nt][ln][32 + hi * 16];
#pragma unroll
    for (int e = 0; e < 16; ++e) Bm1[e] = pb[e];
    pb = &XT[ln][nt][hi * 16];
#pragma unroll
    for (int e = 0; e < 16; ++e) Tm0[e] = pb[e];
    pb = &XT[ln][nt][32 + hi * 16];
#pragma unroll
    for (int e = 0; e < 16; ++e) Tm1[e] = pb[e];

    v8f c = {};
    c = __builtin_amdgcn_wmma_f32_16x16x32_bf16(false, A9a,  false, Bm0, (short)0, c, false, false);
    c = __builtin_amdgcn_wmma_f32_16x16x32_bf16(false, A9b,  false, Bm1, (short)0, c, false, false);
    c = __builtin_amdgcn_wmma_f32_16x16x32_bf16(false, A10a, false, Tm0, (short)0, c, false, false);
    c = __builtin_amdgcn_wmma_f32_16x16x32_bf16(false, A10b, false, Tm1, (short)0, c, false, false);

    const int i = i0 + nt, j = j0 + ln;
    const float dmask = (i == j) ? 1.f : 0.f;  // branchless diagonal select
    const v8f r8 = *reinterpret_cast<const v8f*>(&RS2[nt][sbase]);
    const v8f d8 = *reinterpret_cast<const v8f*>(&DV2[nt][sbase]);
    v8bf o;
#pragma unroll
    for (int r = 0; r < 8; ++r) {
      float v = c[r] + r8[r] + c8[r];
      v = fmaf(d8[r], dmask, v);
      o[r] = (bf16_t)fmaxf(v, 0.f);
    }
    bf16_t* po = Xout + (((size_t)b * M_ + i) * M_ + j) * 64 + sbase;
    *reinterpret_cast<v8bf*>(po) = o;  // 16B store: 8 consecutive channels
  }
}

// ---------------------------------------------------------------------------
// Fused MLP head: y = W2^T relu(W1^T x + b1) + b2, both GEMMs via WMMA,
// h-tile round-trips through LDS in bf16.
// ---------------------------------------------------------------------------
__global__ void __launch_bounds__(256, 1) k_head(
    const bf16_t* __restrict__ X3, const float* __restrict__ w1, const float* __restrict__ b1,
    const float* __restrict__ w2, const float* __restrict__ b2, float* __restrict__ out) {
  const int j0 = blockIdx.x * 16, i0 = blockIdx.y * 16, b = blockIdx.z;
  __shared__ bf16_t W1L[128][64];    // W1L[g][h] = w1[h][g]
  __shared__ bf16_t W2L[16][128];    // W2L[s][g] = w2[g][s], rows 8..15 zero pad
  __shared__ bf16_t XA[16][16][64];  // x tile, channel innermost
  __shared__ bf16_t HT[16][16][128]; // hidden tile, channel innermost
  __shared__ __align__(32) float B1L[128];
  const int t = threadIdx.x;

  for (int idx = t; idx < 128 * 64; idx += 256) {
    int g = idx >> 6, h = idx & 63;
    W1L[g][h] = (bf16_t)w1[(size_t)h * 128 + g];
  }
  for (int idx = t; idx < 16 * 128; idx += 256) {
    int s = idx >> 7, g = idx & 127;
    W2L[s][g] = (s < S_) ? (bf16_t)w2[(size_t)g * S_ + s] : (bf16_t)0.f;
  }
  if (t < 128) B1L[t] = b1[t];
  {
    int ii = t >> 4, jj = t & 15;
    const uint4* s0 = (const uint4*)(X3 + (((size_t)b * M_ + i0 + ii) * M_ + (j0 + jj)) * 64);
    uint4* d0 = (uint4*)&XA[ii][jj][0];
#pragma unroll
    for (int k = 0; k < 8; ++k) d0[k] = s0[k];
  }
  __syncthreads();

  const int wave = t >> 5, lane = t & 31, hi = lane >> 4, ln = lane & 15;

  // Phase 1: h = relu(W1^T x + b1); wave w owns g-block gm = w (16 channels)
  {
    const int gm = wave;
    const int grow = gm * 16 + ln;
    const int gbase = gm * 16 + hi * 8;
    v16bf Aa, Ab;
    const bf16_t* p = &W1L[grow][hi * 16];
#pragma unroll
    for (int e = 0; e < 8; ++e) { Aa[e] = p[e]; Aa[8 + e] = p[8 + e]; }
    p = &W1L[grow][32 + hi * 16];
#pragma unroll
    for (int e = 0; e < 8; ++e) { Ab[e] = p[e]; Ab[8 + e] = p[8 + e]; }
    const v8f bias8 = *reinterpret_cast<const v8f*>(&B1L[gbase]);
    for (int nt = 0; nt < 16; ++nt) {
      v16bf Bm0, Bm1;
      const bf16_t* pb = &XA[nt][ln][hi * 16];
#pragma unroll
      for (int e = 0; e < 16; ++e) Bm0[e] = pb[e];
      pb = &XA[nt][ln][32 + hi * 16];
#pragma unroll
      for (int e = 0; e < 16; ++e) Bm1[e] = pb[e];
      v8f c = {};
      c = __builtin_amdgcn_wmma_f32_16x16x32_bf16(false, Aa, false, Bm0, (short)0, c, false, false);
      c = __builtin_amdgcn_wmma_f32_16x16x32_bf16(false, Ab, false, Bm1, (short)0, c, false, false);
      v8bf o;
#pragma unroll
      for (int r = 0; r < 8; ++r) o[r] = (bf16_t)fmaxf(c[r] + bias8[r], 0.f);
      *reinterpret_cast<v8bf*>(&HT[nt][ln][gbase]) = o;
    }
  }
  __syncthreads();

  // Phase 2: y = W2^T h + b2 (K = 128, 4 WMMAs); waves cover 2 rows each
  {
    v16bf A0, A1, A2, A3;
    const bf16_t* p;
    p = &W2L[ln][0 * 32 + hi * 16];
#pragma unroll
    for (int e = 0; e < 8; ++e) { A0[e] = p[e]; A0[8 + e] = p[8 + e]; }
    p = &W2L[ln][1 * 32 + hi * 16];
#pragma unroll
    for (int e = 0; e < 8; ++e) { A1[e] = p[e]; A1[8 + e] = p[8 + e]; }
    p = &W2L[ln][2 * 32 + hi * 16];
#pragma unroll
    for (int e = 0; e < 8; ++e) { A2[e] = p[e]; A2[8 + e] = p[8 + e]; }
    p = &W2L[ln][3 * 32 + hi * 16];
#pragma unroll
    for (int e = 0; e < 8; ++e) { A3[e] = p[e]; A3[8 + e] = p[8 + e]; }

    for (int nt = wave * 2; nt < wave * 2 + 2; ++nt) {
      v16bf Bm0, Bm1, Bm2, Bm3;
      const bf16_t* pb = &HT[nt][ln][0 * 32 + hi * 16];
#pragma unroll
      for (int e = 0; e < 16; ++e) Bm0[e] = pb[e];
      pb = &HT[nt][ln][1 * 32 + hi * 16];
#pragma unroll
      for (int e = 0; e < 16; ++e) Bm1[e] = pb[e];
      pb = &HT[nt][ln][2 * 32 + hi * 16];
#pragma unroll
      for (int e = 0; e < 16; ++e) Bm2[e] = pb[e];
      pb = &HT[nt][ln][3 * 32 + hi * 16];
#pragma unroll
      for (int e = 0; e < 16; ++e) Bm3[e] = pb[e];

      v8f c = {};
      c = __builtin_amdgcn_wmma_f32_16x16x32_bf16(false, A0, false, Bm0, (short)0, c, false, false);
      c = __builtin_amdgcn_wmma_f32_16x16x32_bf16(false, A1, false, Bm1, (short)0, c, false, false);
      c = __builtin_amdgcn_wmma_f32_16x16x32_bf16(false, A2, false, Bm2, (short)0, c, false, false);
      c = __builtin_amdgcn_wmma_f32_16x16x32_bf16(false, A3, false, Bm3, (short)0, c, false, false);

      if (hi == 0) {  // lanes 0-15 hold s = 0..7 (valid); lanes 16-31 hold pad rows
        const int i = i0 + nt, j = j0 + ln;
#pragma unroll
        for (int r = 0; r < S_; ++r) {
          out[(((size_t)b * S_ + r) * M_ + i) * M_ + j] = c[r] + b2[r];
        }
      }
    }
  }
}

// ---------------------------------------------------------------------------
extern "C" void kernel_launch(void* const* d_in, const int* in_sizes, int n_in,
                              void* d_out, int out_size, void* d_ws, size_t ws_size,
                              hipStream_t stream) {
  const float* P     = (const float*)d_in[0];
  /* d_in[1] = mask: dead code in reference */
  const float* c0    = (const float*)d_in[2];
  const float* bias0 = (const float*)d_in[3];
  const float* c1    = (const float*)d_in[4];
  const float* db1   = (const float*)d_in[5];
  const float* ab1   = (const float*)d_in[6];
  const float* c2    = (const float*)d_in[7];
  const float* db2   = (const float*)d_in[8];
  const float* ab2   = (const float*)d_in[9];
  const float* w1    = (const float*)d_in[10];
  const float* b1    = (const float*)d_in[11];
  const float* w2    = (const float*)d_in[12];
  const float* b2    = (const float*)d_in[13];
  float* y = (float*)d_out;

  char* w = (char*)d_ws;
  auto take = [&](size_t bytes) -> char* {
    char* p = w;
    w += (bytes + 255) & ~size_t(255);
    return p;
  };
  const size_t actBytes = (size_t)B_ * D_ * M_ * M_ * sizeof(bf16_t);  // 151 MB
  bf16_t* actA = (bf16_t*)take(actBytes);
  bf16_t* actB = (bf16_t*)take(actBytes);
  const size_t vecBytes = (size_t)B_ * M_ * D_ * sizeof(float);
  float* rows  = (float*)take(vecBytes);
  float* cols  = (float*)take(vecBytes);
  float* diagv = (float*)take(vecBytes);
  float* rowv  = (float*)take(vecBytes);
  float* colv  = (float*)take(vecBytes);
  float* dvecv = (float*)take(vecBytes);
  const size_t sclBytes = (size_t)B_ * D_ * sizeof(float);
  float* sall  = (float*)take(sclBytes);
  float* sdiag = (float*)take(sclBytes);
  float* smean = (float*)take(sclBytes);
  float* constv = (float*)take(sclBytes);

  dim3 tiles(MT_, MT_, B_);

  // X1 = relu(layer_1_to_2(P))  -> actA (bf16, [b][i][j][h])
  k_meanP<<<B_, D_, 0, stream>>>(P, smean);
  k_proj1<<<B_ * M_, D_, 0, stream>>>(P, smean, c0, bias0, rowv, colv, dvecv, constv);
  k_mat1<<<B_ * M_, 256, 0, stream>>>(rowv, colv, dvecv, constv, actA);

  // X2 = relu(layer_2_to_2(X1))  actA -> actB
  k_rowStats<<<B_ * M_, D_, 0, stream>>>(actA, rows, diagv);
  k_colStats<<<B_ * M_, D_, 0, stream>>>(actA, cols);
  k_scalarStats<<<B_, D_, 0, stream>>>(rows, diagv, sall, sdiag);
  k_proj2<<<B_ * M_, D_, 0, stream>>>(rows, cols, diagv, sdiag, sall, c1, db1, ab1,
                                      rowv, colv, dvecv, constv);
  k_wmma2to2<<<tiles, 256, 0, stream>>>(actA, c1, rowv, colv, dvecv, constv, actB);

  // X3 = relu(layer_2_to_2(X2))  actB -> actA
  k_rowStats<<<B_ * M_, D_, 0, stream>>>(actB, rows, diagv);
  k_colStats<<<B_ * M_, D_, 0, stream>>>(actB, cols);
  k_scalarStats<<<B_, D_, 0, stream>>>(rows, diagv, sall, sdiag);
  k_proj2<<<B_ * M_, D_, 0, stream>>>(rows, cols, diagv, sdiag, sall, c2, db2, ab2,
                                      rowv, colv, dvecv, constv);
  k_wmma2to2<<<tiles, 256, 0, stream>>>(actB, c2, rowv, colv, dvecv, constv, actA);

  // y = MLP head (fused, WMMA)
  k_head<<<tiles, 256, 0, stream>>>(actA, w1, b1, w2, b2, y);
}
